// DAM_42090679500989
// MI455X (gfx1250) — matrix operand order repaired and checked by reading, not compiled, not run on hardware
//
#include <hip/hip_runtime.h>
#include <hip/hip_bf16.h>
#include <math.h>

typedef _Float16 v16h __attribute__((ext_vector_type(16)));
typedef _Float16 v8h  __attribute__((ext_vector_type(8)));
typedef float    v8f  __attribute__((ext_vector_type(8)));

#define TS 7
#define DD 1024
#define HH 1024
#define BB 1024
#define G4 4096  // 4*HH

// ---------------- helpers: WMMA fragment loads (CDNA5 16x16x32 f16) --------

// A fragment: 16x32 tile of row-major A (row stride lda elements).
// lanes 0-15: row M=lane, K = 0..15; lanes 16-31: row M=lane-16, K = 16..31
__device__ __forceinline__ v16h load_a_frag(const _Float16* __restrict__ A,
                                            int lda, int m0, int kk, int lane) {
    int row  = lane & 15;
    int ksel = (lane >> 4) << 4;              // 0 or 16
    const _Float16* p = A + (size_t)(m0 + row) * lda + kk + ksel;
    v8h lo = *(const v8h*)p;                  // K = kbase+0..7
    v8h hi = *(const v8h*)(p + 8);            // K = kbase+8..15
    return __builtin_shufflevector(lo, hi, 0,1,2,3,4,5,6,7,8,9,10,11,12,13,14,15);
}

// B fragment: B = W^T, W row-major [4096 x 1024]. B[k][n] = W[n][k].
// lanes 0-15: N=lane, K=kk+0..15 ; lanes 16-31: N=lane-16, K=kk+16..31
// -> 16 contiguous f16 from one W row (32B aligned).
__device__ __forceinline__ v16h load_b_frag(const _Float16* __restrict__ W,
                                            int n0, int kk, int lane) {
    int nn = n0 + (lane & 15);
    int kb = kk + ((lane >> 4) << 4);         // 0 or 16
    return *(const v16h*)(W + (size_t)nn * 1024 + kb);
}

// ---------------- kernels ---------------------------------------------------

__global__ void f32_to_f16_kernel(const float* __restrict__ in,
                                  _Float16* __restrict__ out, int n) {
    int i = blockIdx.x * blockDim.x + threadIdx.x;
    if (i < n) out[i] = (_Float16)in[i];
}

__global__ void bias_sum_kernel(const float* __restrict__ b_ih,
                                const float* __restrict__ b_hh,
                                float* __restrict__ b2) {
    int i = blockIdx.x * blockDim.x + threadIdx.x;
    if (i < G4) b2[i] = b_ih[i] + b_hh[i];
}

// One block per batch element: cov_temp[7], cov_channel[1024], dist -> f16
__global__ __launch_bounds__(256) void cov_dist_kernel(
    const float* __restrict__ x, const float* __restrict__ xs,
    _Float16* __restrict__ dist) {
    int b   = blockIdx.x;
    int tid = threadIdx.x;
    __shared__ float shSx[TS], shSs[TS], shSxy[TS], covt[TS];
    if (tid < TS) { shSx[tid] = 0.f; shSs[tid] = 0.f; shSxy[tid] = 0.f; }
    __syncthreads();

    const float* xb = x  + (size_t)b * TS * DD;
    const float* sb = xs + (size_t)b * TS * DD;

    float covch[4];
    float rSx[TS], rSs[TS], rSxy[TS];
    #pragma unroll
    for (int t = 0; t < TS; ++t) { rSx[t] = 0.f; rSs[t] = 0.f; rSxy[t] = 0.f; }

    #pragma unroll
    for (int q = 0; q < 4; ++q) {
        int d = tid + (q << 8);
        float cSx = 0.f, cSs = 0.f, cSxy = 0.f;
        #pragma unroll
        for (int t = 0; t < TS; ++t) {
            float xv = xb[t * DD + d];
            float sv = sb[t * DD + d];
            cSx += xv; cSs += sv; cSxy += xv * sv;
            rSx[t] += xv; rSs[t] += sv; rSxy[t] += xv * sv;
        }
        // cov_channel[d] = relu( (Σxy - Σx·Σs/T) / D )
        float cc = (cSxy - cSx * cSs * (1.0f / (float)TS)) * (1.0f / (float)DD);
        covch[q] = cc > 0.f ? cc : 0.f;
    }
    #pragma unroll
    for (int t = 0; t < TS; ++t) {
        atomicAdd(&shSx[t],  rSx[t]);
        atomicAdd(&shSs[t],  rSs[t]);
        atomicAdd(&shSxy[t], rSxy[t]);
    }
    __syncthreads();
    if (tid < TS) {
        // cov_temp[t] = relu( (Σxy - Σx·Σs/D) / T )
        float ct = (shSxy[tid] - shSx[tid] * shSs[tid] * (1.0f / (float)DD))
                   * (1.0f / (float)TS);
        covt[tid] = ct > 0.f ? ct : 0.f;
    }
    __syncthreads();
    #pragma unroll
    for (int q = 0; q < 4; ++q) {
        int d = tid + (q << 8);
        #pragma unroll
        for (int t = 0; t < TS; ++t)
            dist[((size_t)b * TS + t) * DD + d] = (_Float16)(covt[t] * covch[q]);
    }
}

// gates[b][n] = dist_t[b,:]·W_ih[n,:] + h_prev[b,:]·W_hh[n,:] + b2[n]
// grid (64, 8), block 256 (8 waves); wave computes 16 rows x 64 cols.
__global__ __launch_bounds__(256) void gemm_lstm_gates(
    const _Float16* __restrict__ A1, int lda1,      // dist_t, stride 7*1024
    const _Float16* __restrict__ A2, int lda2,      // h16 (null for t=0)
    const _Float16* __restrict__ Wih,
    const _Float16* __restrict__ Whh,
    const float* __restrict__ b2,
    float* __restrict__ out) {                      // [1024 x 4096]
    int wave = threadIdx.x >> 5;
    int lane = threadIdx.x & 31;
    int m0 = blockIdx.x << 4;
    int n0 = (((int)blockIdx.y << 3) + wave) << 6;  // 64-column group

    v8f acc[4];
    #pragma unroll
    for (int i = 0; i < 4; ++i) acc[i] = v8f{};

    for (int kk = 0; kk < 1024; kk += 32) {
        v16h a  = load_a_frag(A1, lda1, m0, kk, lane);
        v16h b0 = load_b_frag(Wih, n0,      kk, lane);
        v16h b1 = load_b_frag(Wih, n0 + 16, kk, lane);
        v16h bb2 = load_b_frag(Wih, n0 + 32, kk, lane);
        v16h b3 = load_b_frag(Wih, n0 + 48, kk, lane);
        acc[0] = __builtin_amdgcn_wmma_f32_16x16x32_f16(false, a, false, b0,  (short)0, acc[0], false, false);
        acc[1] = __builtin_amdgcn_wmma_f32_16x16x32_f16(false, a, false, b1,  (short)0, acc[1], false, false);
        acc[2] = __builtin_amdgcn_wmma_f32_16x16x32_f16(false, a, false, bb2, (short)0, acc[2], false, false);
        acc[3] = __builtin_amdgcn_wmma_f32_16x16x32_f16(false, a, false, b3,  (short)0, acc[3], false, false);
    }
    if (A2) {
        for (int kk = 0; kk < 1024; kk += 32) {
            v16h a  = load_a_frag(A2, lda2, m0, kk, lane);
            v16h b0 = load_b_frag(Whh, n0,      kk, lane);
            v16h b1 = load_b_frag(Whh, n0 + 16, kk, lane);
            v16h bb2 = load_b_frag(Whh, n0 + 32, kk, lane);
            v16h b3 = load_b_frag(Whh, n0 + 48, kk, lane);
            acc[0] = __builtin_amdgcn_wmma_f32_16x16x32_f16(false, a, false, b0,  (short)0, acc[0], false, false);
            acc[1] = __builtin_amdgcn_wmma_f32_16x16x32_f16(false, a, false, b1,  (short)0, acc[1], false, false);
            acc[2] = __builtin_amdgcn_wmma_f32_16x16x32_f16(false, a, false, bb2, (short)0, acc[2], false, false);
            acc[3] = __builtin_amdgcn_wmma_f32_16x16x32_f16(false, a, false, b3,  (short)0, acc[3], false, false);
        }
    }
    // C/D layout: VGPR e -> M = m0 + e + 8*(lane>=16), N = n0 + 16*tile + (lane&15)
    int mrow = m0 + ((lane >> 4) << 3);
    int nlan = lane & 15;
    #pragma unroll
    for (int tile = 0; tile < 4; ++tile) {
        int ncol = n0 + (tile << 4) + nlan;
        float bias = b2[ncol];
        #pragma unroll
        for (int e = 0; e < 8; ++e)
            out[(size_t)(mrow + e) * G4 + ncol] = acc[tile][e] + bias;
    }
}

__device__ __forceinline__ float sigmoidf_(float v) { return 1.0f / (1.0f + expf(-v)); }

__global__ __launch_bounds__(256) void lstm_pointwise(
    const float* __restrict__ gates, float* __restrict__ c,
    float* __restrict__ h_all, _Float16* __restrict__ h16, int t) {
    int idx = blockIdx.x * 256 + threadIdx.x;     // over 1024*1024
    int b = idx >> 10, j = idx & 1023;
    const float* g = gates + (size_t)b * G4;
    float iv = sigmoidf_(g[j]);
    float fv = sigmoidf_(g[HH + j]);
    float gv = tanhf(g[2 * HH + j]);
    float ov = sigmoidf_(g[3 * HH + j]);
    float cp = (t == 0) ? 0.f : c[idx];
    float cn = fv * cp + iv * gv;
    c[idx] = cn;
    float h = ov * tanhf(cn);
    h_all[((size_t)b * TS + t) * HH + j] = h;
    h16[idx] = (_Float16)h;
}

__global__ __launch_bounds__(256) void softmax_time_kernel(
    const float* __restrict__ h_all, float* __restrict__ out) {
    int idx = blockIdx.x * 256 + threadIdx.x;     // over 1024*1024
    int b = idx >> 10, j = idx & 1023;
    const float* hp = h_all + (size_t)b * TS * HH + j;
    float v[TS], m = -3.0e38f;
    #pragma unroll
    for (int t = 0; t < TS; ++t) { v[t] = hp[t * HH]; m = fmaxf(m, v[t]); }
    float s = 0.f;
    #pragma unroll
    for (int t = 0; t < TS; ++t) { v[t] = expf(v[t] - m); s += v[t]; }
    float inv = 1.0f / s;
    #pragma unroll
    for (int t = 0; t < TS; ++t)
        out[((size_t)b * TS + t) * HH + j] = v[t] * inv;
}

// one wave per (b,t): sigmoid( Σ_j tanh(h[b,t,j]) * attn_W[j] + attn_b )
__global__ __launch_bounds__(256) void temporal_attn_kernel(
    const float* __restrict__ h_all, const float* __restrict__ aw,
    const float* __restrict__ ab, float* __restrict__ out) {
    int gw   = (blockIdx.x * 256 + threadIdx.x) >> 5;   // 0..7167
    int lane = threadIdx.x & 31;
    const float* hp = h_all + (size_t)gw * HH;
    float s = 0.f;
    #pragma unroll
    for (int i = 0; i < 32; ++i) {
        int j = lane + (i << 5);
        s += tanhf(hp[j]) * aw[j];
    }
    #pragma unroll
    for (int m = 16; m > 0; m >>= 1) s += __shfl_xor(s, m, 32);
    if (lane == 0) out[gw] = 1.0f / (1.0f + expf(-(s + ab[0])));
}

// ---------------- launch ----------------------------------------------------

extern "C" void kernel_launch(void* const* d_in, const int* in_sizes, int n_in,
                              void* d_out, int out_size, void* d_ws, size_t ws_size,
                              hipStream_t stream) {
    const float* x      = (const float*)d_in[0];
    const float* xs     = (const float*)d_in[1];
    const float* W_ih   = (const float*)d_in[2];
    const float* W_hh   = (const float*)d_in[3];
    const float* b_ih   = (const float*)d_in[4];
    const float* b_hh   = (const float*)d_in[5];
    const float* attn_W = (const float*)d_in[6];
    const float* attn_b = (const float*)d_in[7];
    float* out = (float*)d_out;

    char* ws = (char*)d_ws;
    size_t off = 0;
    auto alloc = [&](size_t bytes) -> void* {
        void* p = ws + off; off += (bytes + 255) & ~(size_t)255; return p;
    };
    _Float16* W16i   = (_Float16*)alloc((size_t)G4 * DD * 2);   //  8 MB
    _Float16* W16h   = (_Float16*)alloc((size_t)G4 * HH * 2);   //  8 MB
    float*    b2     = (float*)   alloc((size_t)G4 * 4);        // 16 KB
    _Float16* dist16 = (_Float16*)alloc((size_t)BB * TS * DD * 2); // 14 MB
    float*    gates  = (float*)   alloc((size_t)BB * G4 * 4);   // 16 MB
    float*    h_all  = (float*)   alloc((size_t)BB * TS * HH * 4); // 28 MB
    _Float16* h16    = (_Float16*)alloc((size_t)BB * HH * 2);   //  2 MB
    float*    cbuf   = (float*)   alloc((size_t)BB * HH * 4);   //  4 MB

    // 1. weight conversion + bias sum
    f32_to_f16_kernel<<<(G4 * DD) / 256, 256, 0, stream>>>(W_ih, W16i, G4 * DD);
    f32_to_f16_kernel<<<(G4 * HH) / 256, 256, 0, stream>>>(W_hh, W16h, G4 * HH);
    bias_sum_kernel<<<G4 / 256, 256, 0, stream>>>(b_ih, b_hh, b2);

    // 2. covariance outer product -> dist (f16)
    cov_dist_kernel<<<BB, 256, 0, stream>>>(x, xs, dist16);

    // 3. LSTM: fused (dist_t·W_ih^T + h·W_hh^T + b2) GEMM, then gate pointwise
    for (int t = 0; t < TS; ++t) {
        gemm_lstm_gates<<<dim3(BB / 16, 8), 256, 0, stream>>>(
            dist16 + (size_t)t * DD, TS * DD,
            (t == 0) ? (const _Float16*)nullptr : h16, HH,
            W16i, W16h, b2, gates);
        lstm_pointwise<<<(BB * HH) / 256, 256, 0, stream>>>(gates, cbuf, h_all, h16, t);
    }

    // 4. attention outputs
    softmax_time_kernel<<<(BB * HH) / 256, 256, 0, stream>>>(h_all, out);
    temporal_attn_kernel<<<(BB * TS) / 8, 256, 0, stream>>>(
        h_all, attn_W, attn_b, out + (size_t)BB * TS * HH);
}
